// AttnHead_81028853006993
// MI455X (gfx1250) — matrix operand order repaired and checked by reading, not compiled, not run on hardware
//
#include <hip/hip_runtime.h>
#include <hip/hip_bf16.h>
#include <stdint.h>

// Problem constants
#define BB     8
#define TSEQ   2048
#define NEMB   512
#define NHEAD  4
#define HD     128
#define NROWS  (BB * TSEQ)          // 16384 token rows
#define TW     (TSEQ / 32)          // 64 mask words per query row

typedef __attribute__((ext_vector_type(16))) _Float16 v16h;
typedef __attribute__((ext_vector_type(8)))  _Float16 h8;   // 16-byte load unit
typedef __attribute__((ext_vector_type(8)))  float    v8f;

// ---------------------------------------------------------------------------
// WMMA fragment loaders.  Source is row-major [outer, K] in f16.
// A matrix: lane holds row M=lane%16; per 32-K chunk: K = half*8+{0..7} and
//           K = 16+half*8+{0..7}.
// B matrix (fed as B^T rows): lane holds col N=lane%16; per 32-K chunk:
//           K = half*8+{0..7} and K = 16+half*8+{0..7}.
// ---------------------------------------------------------------------------
__device__ __forceinline__ v16h load_frag_a(const _Float16* row, int kbase, int half) {
  h8 lo = *(const h8*)(row + kbase + half * 8);
  h8 hi = *(const h8*)(row + kbase + 16 + half * 8);
  v16h r;
#pragma unroll
  for (int i = 0; i < 8; ++i) { r[i] = lo[i]; r[i + 8] = hi[i]; }
  return r;
}

__device__ __forceinline__ v16h load_frag_b(const _Float16* rowT, int kbase, int half) {
  h8 lo = *(const h8*)(rowT + kbase + half * 8);
  h8 hi = *(const h8*)(rowT + kbase + 16 + half * 8);
  v16h r;
#pragma unroll
  for (int i = 0; i < 8; ++i) { r[i] = lo[i]; r[i + 8] = hi[i]; }
  return r;
}

__device__ __forceinline__ v8f wmma16(v16h a, v16h b, v8f c) {
  return __builtin_amdgcn_wmma_f32_16x16x32_f16(false, a, false, b, (short)0, c,
                                                false, false);
}

// CDNA5 async copy: per-lane 16B global -> LDS, tracked by ASYNCcnt.
__device__ __forceinline__ void async_b128(uint32_t lds_off, uint64_t gaddr) {
  asm volatile("global_load_async_to_lds_b128 %0, %1, off"
               :: "v"(lds_off), "v"(gaddr) : "memory");
}
__device__ __forceinline__ void wait_async0() {
  asm volatile("s_wait_asynccnt 0" ::: "memory");
}

// ---------------------------------------------------------------------------
// 1) LayerNorm + extra affine, fp32 -> f16.  One wave per 512-wide row.
// ---------------------------------------------------------------------------
__global__ __launch_bounds__(128) void ln_kernel(
    const float* __restrict__ x, const float* __restrict__ lnw,
    const float* __restrict__ lnb, const float* __restrict__ affw,
    const float* __restrict__ affb, _Float16* __restrict__ xn) {
  int widx = threadIdx.x >> 5, lane = threadIdx.x & 31;
  int row = blockIdx.x * 4 + widx;
  const float* xr = x + (size_t)row * NEMB;
  float tv[16];
  float s = 0.f, ss = 0.f;
#pragma unroll
  for (int i = 0; i < 4; ++i) {
    float4 v = ((const float4*)xr)[lane * 4 + i];
    tv[4 * i + 0] = v.x; tv[4 * i + 1] = v.y;
    tv[4 * i + 2] = v.z; tv[4 * i + 3] = v.w;
    s  += v.x + v.y + v.z + v.w;
    ss += v.x * v.x + v.y * v.y + v.z * v.z + v.w * v.w;
  }
#pragma unroll
  for (int m = 1; m < 32; m <<= 1) {
    s  += __shfl_xor(s, m, 32);
    ss += __shfl_xor(ss, m, 32);
  }
  float mean = s * (1.f / NEMB);
  float var  = ss * (1.f / NEMB) - mean * mean;
  float rstd = rsqrtf(var + 1e-5f);
  h8 o0, o1;
#pragma unroll
  for (int j = 0; j < 16; ++j) {
    int c = lane * 16 + j;
    float y = (tv[j] - mean) * rstd;
    y = y * lnw[c] + lnb[c];
    y = y * affw[c] + affb[c];
    if (j < 8) o0[j] = (_Float16)y; else o1[j - 8] = (_Float16)y;
  }
  _Float16* orow = xn + (size_t)row * NEMB + lane * 16;
  *(h8*)(orow)     = o0;
  *(h8*)(orow + 8) = o1;
}

// ---------------------------------------------------------------------------
// 2) Transpose fp32 weight [512, N] -> f16 [N, 512]
// ---------------------------------------------------------------------------
__global__ __launch_bounds__(256) void transpose_kernel(
    const float* __restrict__ in, _Float16* __restrict__ out, int N) {
  size_t id = (size_t)blockIdx.x * 256 + threadIdx.x;  // over N*512
  int n = (int)(id >> 9);
  int c = (int)(id & 511);
  out[id] = (_Float16)in[(size_t)c * N + n];
}

// ---------------------------------------------------------------------------
// 3) Pack mask_table[pos_emb] into bit words via wave32 ballot.
// ---------------------------------------------------------------------------
__global__ __launch_bounds__(256) void maskpack_kernel(
    const int* __restrict__ pos, const unsigned char* __restrict__ table,
    unsigned int* __restrict__ bits) {
  size_t id = (size_t)blockIdx.x * 256 + threadIdx.x;  // over T*T
  int idx = pos[id];
  bool vis = table[idx] != 0;
  unsigned int bal = __builtin_amdgcn_ballot_w32(vis);
  if ((threadIdx.x & 31) == 0) bits[id >> 5] = bal;
}

// ---------------------------------------------------------------------------
// 4) QKV GEMM: one wave computes a 32x64 tile, register double-buffered.
//    Scatters q/k [B,H,T,128] f16 and V transposed [B,H,128,T] f16.
// ---------------------------------------------------------------------------
__global__ __launch_bounds__(128) void qkv_gemm_kernel(
    const _Float16* __restrict__ A, const _Float16* __restrict__ Bt,
    _Float16* __restrict__ qo, _Float16* __restrict__ ko,
    _Float16* __restrict__ vto) {
  int widx = threadIdx.x >> 5, lane = threadIdx.x & 31;
  int col = lane & 15, half = lane >> 4;
  int w = blockIdx.x * 4 + widx;           // 12288 waves
  int mt = w / 24, strip = w - mt * 24;
  int m0 = mt << 5, n0 = strip << 6;

  const _Float16* arow0 = A + (size_t)(m0 + col) * NEMB;
  const _Float16* arow1 = arow0 + (size_t)16 * NEMB;
  const _Float16* brow[4];
#pragma unroll
  for (int nt = 0; nt < 4; ++nt)
    brow[nt] = Bt + (size_t)(n0 + nt * 16 + col) * NEMB;

  v8f acc[8];
#pragma unroll
  for (int i = 0; i < 8; ++i) acc[i] = (v8f){};

  v16h ca0 = load_frag_a(arow0, 0, half);
  v16h ca1 = load_frag_a(arow1, 0, half);
  v16h cb0 = load_frag_b(brow[0], 0, half);
  v16h cb1 = load_frag_b(brow[1], 0, half);
  v16h cb2 = load_frag_b(brow[2], 0, half);
  v16h cb3 = load_frag_b(brow[3], 0, half);

#pragma unroll
  for (int k0 = 0; k0 < NEMB; k0 += 32) {
    int kn = k0 + 32;
    v16h na0 = ca0, na1 = ca1, nb0 = cb0, nb1 = cb1, nb2 = cb2, nb3 = cb3;
    if (kn < NEMB) {           // uniform, folds at compile time under unroll
      na0 = load_frag_a(arow0, kn, half);
      na1 = load_frag_a(arow1, kn, half);
      nb0 = load_frag_b(brow[0], kn, half);
      nb1 = load_frag_b(brow[1], kn, half);
      nb2 = load_frag_b(brow[2], kn, half);
      nb3 = load_frag_b(brow[3], kn, half);
    }
    acc[0] = wmma16(ca0, cb0, acc[0]);
    acc[1] = wmma16(ca0, cb1, acc[1]);
    acc[2] = wmma16(ca0, cb2, acc[2]);
    acc[3] = wmma16(ca0, cb3, acc[3]);
    acc[4] = wmma16(ca1, cb0, acc[4]);
    acc[5] = wmma16(ca1, cb1, acc[5]);
    acc[6] = wmma16(ca1, cb2, acc[6]);
    acc[7] = wmma16(ca1, cb3, acc[7]);
    ca0 = na0; ca1 = na1; cb0 = nb0; cb1 = nb1; cb2 = nb2; cb3 = nb3;
  }

#pragma unroll
  for (int nt = 0; nt < 4; ++nt) {
    int n = n0 + nt * 16 + col;
    int part = n >> 9;                 // 0=q,1=k,2=v
    int within = n & 511;
    int hh = within >> 7, d = within & 127;
#pragma unroll
    for (int i = 0; i < 2; ++i) {
#pragma unroll
      for (int r = 0; r < 8; ++r) {
        int m = m0 + i * 16 + half * 8 + r;
        int b = m >> 11, t = m & 2047;
        size_t bh = (size_t)(b * NHEAD + hh);
        _Float16 v = (_Float16)acc[i * 4 + nt][r];
        if (part == 0)      qo[(bh * TSEQ + t) * HD + d] = v;
        else if (part == 1) ko[(bh * TSEQ + t) * HD + d] = v;
        else                vto[(bh * HD + d) * TSEQ + t] = v;
      }
    }
  }
}

// ---------------------------------------------------------------------------
// 5) Flash attention.  Block = 4 waves sharing one (b,h); each wave owns a
//    16-query tile.  K(32x128) and V^T(128x32) tiles are staged into LDS with
//    double-buffered async copies (ASYNCcnt) so DMA of step j+1 runs under
//    the WMMAs/softmax of step j.
// ---------------------------------------------------------------------------
__global__ __launch_bounds__(128) void attn_kernel(
    const _Float16* __restrict__ q, const _Float16* __restrict__ k,
    const _Float16* __restrict__ vt, const unsigned int* __restrict__ mbits,
    _Float16* __restrict__ o) {
  __shared__ _Float16 kbuf[2][32 * HD];   // 2 x 8KB
  __shared__ _Float16 vbuf[2][HD * 32];   // 2 x 8KB
  __shared__ _Float16 pbuf[4][16 * 32];   // 4KB

  const int tid  = threadIdx.x;
  const int widx = tid >> 5, lane = tid & 31;
  const int col  = lane & 15, half = lane >> 4;
  const int bh   = blockIdx.x >> 5;                  // 0..31 = b*4+h
  const int qt   = ((blockIdx.x & 31) << 2) | widx;  // 0..127
  const int q0   = qt << 4;
  const float scale = 0.0883883476483184f;           // 1/sqrt(128)

  const char* kbase_g = (const char*)(k + (size_t)bh * TSEQ * HD);
  const char* vrow_g  = (const char*)(vt + ((size_t)bh * HD + tid) * TSEQ);

  // stage step-j tiles into buffer p (8 async b128 per wave)
  auto stage = [&](int j, int p) {
    const uint32_t lk = (uint32_t)(uintptr_t)&kbuf[p][0];
    const char* gk = kbase_g + ((size_t)j << 5) * (HD * 2);  // contiguous 8KB
#pragma unroll
    for (int i = 0; i < 4; ++i)
      async_b128(lk + (uint32_t)(tid + 128 * i) * 16,
                 (uint64_t)(uintptr_t)(gk + (size_t)(tid + 128 * i) * 16));
    const uint32_t lv = (uint32_t)(uintptr_t)&vbuf[p][tid * 32];
    const char* gv = vrow_g + ((size_t)j << 5) * 2;          // 64B per row
#pragma unroll
    for (int c = 0; c < 4; ++c)
      async_b128(lv + c * 16, (uint64_t)(uintptr_t)(gv + c * 16));
  };

  stage(0, 0);

  const _Float16* qbase = q + ((size_t)bh * TSEQ + q0 + col) * HD;
  v16h qf[4];
#pragma unroll
  for (int c = 0; c < 4; ++c) qf[c] = load_frag_a(qbase, c * 32, half);

  float mrow[8], lrow[8];
  v8f acc[8];
#pragma unroll
  for (int r = 0; r < 8; ++r) { mrow[r] = -1e30f; lrow[r] = 0.f; }
#pragma unroll
  for (int nt = 0; nt < 8; ++nt) acc[nt] = (v8f){};

  _Float16* pl = pbuf[widx];

  for (int j = 0; j < TW; ++j) {
    const int p = j & 1;
    wait_async0();        // our buffer-p copies (issued last step) are done
    __syncthreads();      // everyone's copies done; previous reads retired
    if (j + 1 < TW) stage(j + 1, p ^ 1);
    __builtin_prefetch(&mbits[(size_t)q0 * TW + j + 1], 0, 0);

    // ---- scores: two 16x16 tiles over 32 keys from LDS, K-dim = 128 ----
    const _Float16* kp0 = &kbuf[p][col * HD];
    const _Float16* kp1 = &kbuf[p][(16 + col) * HD];
    v8f s0 = (v8f){}, s1 = (v8f){};
#pragma unroll
    for (int c = 0; c < 4; ++c) {
      s0 = wmma16(qf[c], load_frag_b(kp0, c * 32, half), s0);
      s1 = wmma16(qf[c], load_frag_b(kp1, c * 32, half), s1);
    }
    // ---- mask words ----
    unsigned int wrd[8];
#pragma unroll
    for (int r = 0; r < 8; ++r)
      wrd[r] = mbits[(size_t)(q0 + half * 8 + r) * TW + j];

    // ---- online softmax update ----
    float p0[8], p1[8], alpha[8];
#pragma unroll
    for (int r = 0; r < 8; ++r) {
      float a0 = s0[r] * scale;
      float a1 = s1[r] * scale;
      if (!((wrd[r] >> col) & 1u))        a0 = -1e9f;
      if (!((wrd[r] >> (16 + col)) & 1u)) a1 = -1e9f;
      float mx = fmaxf(a0, a1);
#pragma unroll
      for (int m = 1; m < 16; m <<= 1) mx = fmaxf(mx, __shfl_xor(mx, m, 32));
      float mn = fmaxf(mrow[r], mx);
      float al = __expf(mrow[r] - mn);
      p0[r] = __expf(a0 - mn);
      p1[r] = __expf(a1 - mn);
      float rs = p0[r] + p1[r];
#pragma unroll
      for (int m = 1; m < 16; m <<= 1) rs += __shfl_xor(rs, m, 32);
      lrow[r] = lrow[r] * al + rs;
      mrow[r] = mn;
      alpha[r] = al;
    }
#pragma unroll
    for (int nt = 0; nt < 8; ++nt)
#pragma unroll
      for (int r = 0; r < 8; ++r) acc[nt][r] *= alpha[r];

    // ---- stage P (16x32 row-major) through LDS, reload as A fragment ----
    __builtin_amdgcn_wave_barrier();
#pragma unroll
    for (int r = 0; r < 8; ++r) {
      pl[(half * 8 + r) * 32 + col]      = (_Float16)p0[r];
      pl[(half * 8 + r) * 32 + 16 + col] = (_Float16)p1[r];
    }
    __builtin_amdgcn_wave_barrier();
    v16h pf = load_frag_a(pl + col * 32, 0, half);
    __builtin_amdgcn_wave_barrier();

    // ---- accumulate P @ V  (V^T rows from LDS as B operand) ----
#pragma unroll
    for (int nt = 0; nt < 8; ++nt) {
      const _Float16* vrow = &vbuf[p][(nt * 16 + col) * 32];
      acc[nt] = wmma16(pf, load_frag_b(vrow, 0, half), acc[nt]);
    }
  }

  // ---- epilogue: normalize and write [B, T, 512] f16 row-major ----
  int b = bh >> 2, h = bh & 3;
#pragma unroll
  for (int r = 0; r < 8; ++r) {
    float inv = 1.f / lrow[r];
    int m = q0 + half * 8 + r;
    _Float16* orow = o + ((size_t)b * TSEQ + m) * NEMB + h * HD;
#pragma unroll
    for (int nt = 0; nt < 8; ++nt)
      orow[nt * 16 + col] = (_Float16)(acc[nt][r] * inv);
  }
}

// ---------------------------------------------------------------------------
// 6) Output projection: 32x64 tile per wave, register double-buffered.
// ---------------------------------------------------------------------------
__global__ __launch_bounds__(128) void proj_gemm_kernel(
    const _Float16* __restrict__ A, const _Float16* __restrict__ Bt,
    float* __restrict__ out) {
  int widx = threadIdx.x >> 5, lane = threadIdx.x & 31;
  int col = lane & 15, half = lane >> 4;
  int w = blockIdx.x * 4 + widx;           // 4096 waves
  int mt = w >> 3, strip = w & 7;
  int m0 = mt << 5, n0 = strip << 6;

  const _Float16* arow0 = A + (size_t)(m0 + col) * NEMB;
  const _Float16* arow1 = arow0 + (size_t)16 * NEMB;
  const _Float16* brow[4];
#pragma unroll
  for (int nt = 0; nt < 4; ++nt)
    brow[nt] = Bt + (size_t)(n0 + nt * 16 + col) * NEMB;

  v8f acc[8];
#pragma unroll
  for (int i = 0; i < 8; ++i) acc[i] = (v8f){};

  v16h ca0 = load_frag_a(arow0, 0, half);
  v16h ca1 = load_frag_a(arow1, 0, half);
  v16h cb0 = load_frag_b(brow[0], 0, half);
  v16h cb1 = load_frag_b(brow[1], 0, half);
  v16h cb2 = load_frag_b(brow[2], 0, half);
  v16h cb3 = load_frag_b(brow[3], 0, half);

#pragma unroll
  for (int k0 = 0; k0 < NEMB; k0 += 32) {
    int kn = k0 + 32;
    v16h na0 = ca0, na1 = ca1, nb0 = cb0, nb1 = cb1, nb2 = cb2, nb3 = cb3;
    if (kn < NEMB) {
      na0 = load_frag_a(arow0, kn, half);
      na1 = load_frag_a(arow1, kn, half);
      nb0 = load_frag_b(brow[0], kn, half);
      nb1 = load_frag_b(brow[1], kn, half);
      nb2 = load_frag_b(brow[2], kn, half);
      nb3 = load_frag_b(brow[3], kn, half);
    }
    acc[0] = wmma16(ca0, cb0, acc[0]);
    acc[1] = wmma16(ca0, cb1, acc[1]);
    acc[2] = wmma16(ca0, cb2, acc[2]);
    acc[3] = wmma16(ca0, cb3, acc[3]);
    acc[4] = wmma16(ca1, cb0, acc[4]);
    acc[5] = wmma16(ca1, cb1, acc[5]);
    acc[6] = wmma16(ca1, cb2, acc[6]);
    acc[7] = wmma16(ca1, cb3, acc[7]);
    ca0 = na0; ca1 = na1; cb0 = nb0; cb1 = nb1; cb2 = nb2; cb3 = nb3;
  }

#pragma unroll
  for (int nt = 0; nt < 4; ++nt) {
    int n = n0 + nt * 16 + col;
#pragma unroll
    for (int i = 0; i < 2; ++i) {
#pragma unroll
      for (int r = 0; r < 8; ++r) {
        int m = m0 + i * 16 + half * 8 + r;
        out[(size_t)m * NEMB + n] = acc[i * 4 + nt][r];
      }
    }
  }
}

// ---------------------------------------------------------------------------
// Launch
// ---------------------------------------------------------------------------
extern "C" void kernel_launch(void* const* d_in, const int* in_sizes, int n_in,
                              void* d_out, int out_size, void* d_ws, size_t ws_size,
                              hipStream_t stream) {
  (void)in_sizes; (void)n_in; (void)out_size; (void)ws_size;
  const float*         x    = (const float*)d_in[0];
  const int*           pos  = (const int*)d_in[1];
  const float*         lnw  = (const float*)d_in[2];
  const float*         lnb  = (const float*)d_in[3];
  const float*         affw = (const float*)d_in[4];
  const float*         affb = (const float*)d_in[5];
  const float*         wqkv = (const float*)d_in[6];
  const unsigned char* mtab = (const unsigned char*)d_in[7];
  const float*         wprj = (const float*)d_in[8];
  float* out = (float*)d_out;

  char* ws = (char*)d_ws;
  _Float16*     xn     = (_Float16*)(ws + 0);            // 16 MB; reused as attn out
  _Float16*     wqkvT  = (_Float16*)(ws + 16777216);     // 1.5 MB
  _Float16*     wprojT = (_Float16*)(ws + 18350080);     // 0.5 MB
  unsigned int* mbits  = (unsigned int*)(ws + 18874368); // 0.5 MB
  _Float16*     qh     = (_Float16*)(ws + 19398656);     // 16 MB
  _Float16*     kh     = (_Float16*)(ws + 36175872);     // 16 MB
  _Float16*     vth    = (_Float16*)(ws + 52953088);     // 16 MB  (total ~66.5 MB)

  ln_kernel<<<dim3(NROWS / 4), dim3(128), 0, stream>>>(x, lnw, lnb, affw, affb, xn);
  transpose_kernel<<<dim3((1536 * 512) / 256), dim3(256), 0, stream>>>(wqkv, wqkvT, 1536);
  transpose_kernel<<<dim3((512 * 512) / 256), dim3(256), 0, stream>>>(wprj, wprojT, 512);
  maskpack_kernel<<<dim3((TSEQ * TSEQ) / 256), dim3(256), 0, stream>>>(pos, mtab, mbits);
  qkv_gemm_kernel<<<dim3((512 * 24) / 4), dim3(128), 0, stream>>>(xn, wqkvT, qh, kh, vth);
  attn_kernel<<<dim3((32 * 32)), dim3(128), 0, stream>>>(qh, kh, vth, mbits, xn /* o */);
  proj_gemm_kernel<<<dim3((512 * 8) / 4), dim3(128), 0, stream>>>(xn, wprojT, out);
}
